// PaddedPipeline_15358803051123
// MI455X (gfx1250) — matrix operand order repaired and checked by reading, not compiled, not run on hardware
//
#include <hip/hip_runtime.h>
#include <math.h>

// ---------------------------------------------------------------------------
// ViT-B/16 forward with token pruning (B=64), CDNA5 / gfx1250, wave32 WMMA.
// All matmuls via v_wmma_f32_16x16x32_bf16; GEMM tiles staged to LDS with
// global_load_async_to_lds_b128 (double-buffered, s_wait_asynccnt).
// GEMM: 256 thr / 8 waves, block tile 128x128, wave tile 32x64 -> ~190 VGPRs
// peak even with fully-hoisted fragment loads (no scratch spills).
// ---------------------------------------------------------------------------

#define BT 64
#define DIM 768
#define NH 12
#define HD 64
#define DFF 3072
#define NLAYER 12
#define SREAL 197
#define SP1 224          // dense phase padded tokens (multiple of 32)
#define SP2 128          // pruned phase padded tokens (multiple of 32)
#define NKEEP 98
#define NPATCH 196
#define NCLS 1000
#define NEGM (-1.0e30f)

typedef __attribute__((ext_vector_type(16))) __bf16 v16bf;
typedef __attribute__((ext_vector_type(8)))  float  v8f;
typedef __attribute__((ext_vector_type(4)))  unsigned u32x4;
typedef __attribute__((ext_vector_type(4)))  float  f32x4;

union FragU { v16bf f; u32x4 u[2]; };

__device__ __forceinline__ unsigned f2bf(float x) {
  unsigned u = __float_as_uint(x);
  return ((u + 0x7fffu + ((u >> 16) & 1u)) >> 16) & 0xffffu;   // RNE
}

__device__ __forceinline__ v8f wmma_bf16(const FragU& a, const FragU& b, v8f c) {
  return __builtin_amdgcn_wmma_f32_16x16x32_bf16(false, a.f, false, b.f,
                                                 (short)0, c, false, false);
}

// CDNA5 async copy: 16B per lane, global -> LDS, tracked by ASYNCcnt.
__device__ __forceinline__ void async_copy_b128(unsigned lds_off, const void* gaddr) {
  asm volatile("global_load_async_to_lds_b128 %0, %1, off"
               :: "v"(lds_off), "v"((unsigned long long)(size_t)gaddr)
               : "memory");
}
__device__ __forceinline__ void async_wait0() {
  asm volatile("s_wait_asynccnt 0x0" ::: "memory");
}
__device__ __forceinline__ void async_wait4() {   // allow next tile's 4 in flight
  asm volatile("s_wait_asynccnt 0x4" ::: "memory");
}

enum { MODE_BF16 = 0, MODE_GELU = 1, MODE_RESF32 = 2, MODE_QKV = 3, MODE_PATCH = 4 };

// f32 -> bf16 bulk weight conversion (once per launch)
__global__ void k_cvt(const float* __restrict__ s, unsigned short* __restrict__ d, int n) {
  int i = blockIdx.x * 256 + threadIdx.x;
  if (i < n) d[i] = (unsigned short)f2bf(s[i]);
}

// ---------------------------------------------------------------------------
// GEMM: out[m,n] = sum_k A[m,k]*W[n,k] + bias[n]   (out = A @ W^T)
// A,W bf16 global. Block = 256 thr (8 waves, 4x2), tile 128x128,
// wave tile 32x64, K-step 32. Interior blocks: double-buffered async LDS
// staging; edge blocks: guarded synchronous staging.
// ---------------------------------------------------------------------------
__global__ void __launch_bounds__(256, 1)
k_gemm(const unsigned short* __restrict__ A,
       const unsigned short* __restrict__ Wb,
       const float* __restrict__ bias,
       int M, int N, int K, int mode,
       float* __restrict__ outF,
       unsigned short* __restrict__ outB, int ldo,
       const float* __restrict__ resid,
       const float* __restrict__ pos,
       unsigned short* __restrict__ qb,
       unsigned short* __restrict__ kbf,
       unsigned short* __restrict__ vtb,
       int Sp) {
  __shared__ __align__(16) unsigned short sA[2][128 * 40];
  __shared__ __align__(16) unsigned short sW[2][128 * 40];

  const int tid  = threadIdx.x;
  const int lane = tid & 31;
  const int wave = tid >> 5;         // 0..7
  const int lh   = lane >> 4;        // lane half (0/1)
  const int l16  = lane & 15;
  const int wm   = (wave & 3) * 32;  // 4 M-tiles of 32
  const int wn   = (wave >> 2) * 64; // 2 N-tiles of 64
  const int mBase = blockIdx.y * 128;
  const int nBase = blockIdx.x * 128;

  const v8f vzero = {0.f,0.f,0.f,0.f,0.f,0.f,0.f,0.f};
  v8f acc[2][4];
#pragma unroll
  for (int i = 0; i < 2; i++)
#pragma unroll
    for (int j = 0; j < 4; j++) acc[i][j] = vzero;

  // issue 4 async b128 copies per thread (2 for A tile, 2 for W tile)
  auto stage_async = [&](int buf, int kb) {
#pragma unroll
    for (int it = 0; it < 2; it++) {
      int idx = tid + it * 256;              // 0..511: r = idx/4, ch = idx%4
      int r = idx >> 2, ch = idx & 3;
      async_copy_b128((unsigned)(size_t)&sA[buf][r * 40 + ch * 8],
                      (const void*)(A + (size_t)(mBase + r) * K + kb + ch * 8));
    }
#pragma unroll
    for (int it = 0; it < 2; it++) {
      int idx = tid + it * 256;
      int r = idx >> 2, ch = idx & 3;
      async_copy_b128((unsigned)(size_t)&sW[buf][r * 40 + ch * 8],
                      (const void*)(Wb + (size_t)(nBase + r) * K + kb + ch * 8));
    }
  };

  auto stage_sync = [&](int kb) {
#pragma unroll
    for (int it = 0; it < 2; it++) {
      int idx = tid + it * 256;
      int r = idx >> 2, ch = idx & 3;
      u32x4 va = {0u, 0u, 0u, 0u};
      if (mBase + r < M) va = *(const u32x4*)(A + (size_t)(mBase + r) * K + kb + ch * 8);
      *(u32x4*)&sA[0][r * 40 + ch * 8] = va;
      u32x4 vw = {0u, 0u, 0u, 0u};
      if (nBase + r < N) vw = *(const u32x4*)(Wb + (size_t)(nBase + r) * K + kb + ch * 8);
      *(u32x4*)&sW[0][r * 40 + ch * 8] = vw;
    }
  };

  auto compute = [&](int buf) {
    FragU af[2];
#pragma unroll
    for (int i = 0; i < 2; i++) {       // A frag: K = 16*(e>=8) + 8*lh + e%8
      int row = wm + i * 16 + l16;
      af[i].u[0] = *(const u32x4*)&sA[buf][row * 40 + lh * 8];
      af[i].u[1] = *(const u32x4*)&sA[buf][row * 40 + 16 + lh * 8];
    }
#pragma unroll
    for (int j = 0; j < 4; j++) {       // B frag: K = 16*(e>=8) + 8*lh + e%8
      int col = wn + j * 16 + l16;
      FragU bf;
      bf.u[0] = *(const u32x4*)&sW[buf][col * 40 + lh * 8];
      bf.u[1] = *(const u32x4*)&sW[buf][col * 40 + 16 + lh * 8];
#pragma unroll
      for (int i = 0; i < 2; i++) acc[i][j] = wmma_bf16(af[i], bf, acc[i][j]);
    }
  };

  const bool interior = (mBase + 128 <= M) && (nBase + 128 <= N);
  if (interior) {
    stage_async(0, 0);
    int p = 0;
    for (int kb = 0; kb < K; kb += 32) {
      if (kb + 32 < K) { stage_async(p ^ 1, kb + 32); async_wait4(); }
      else             { async_wait0(); }
      __syncthreads();            // tile p visible to all waves
      compute(p);
      __syncthreads();            // all reads of tile p done before restage
      p ^= 1;
    }
  } else {
    for (int kb = 0; kb < K; kb += 32) {
      __syncthreads();
      stage_sync(kb);
      __syncthreads();
      compute(0);
    }
  }

  // epilogue. C layout: row = r + 8*lh (within 16-tile), col = l16.
#pragma unroll
  for (int i = 0; i < 2; i++)
#pragma unroll
    for (int j = 0; j < 4; j++)
#pragma unroll
      for (int r = 0; r < 8; r++) {
        int m = mBase + wm + i * 16 + lh * 8 + r;
        int n = nBase + wn + j * 16 + l16;
        if (m >= M || n >= N) continue;
        float v = acc[i][j][r] + (bias ? bias[n] : 0.f);
        if (mode == MODE_BF16) {
          outB[(size_t)m * ldo + n] = (unsigned short)f2bf(v);
        } else if (mode == MODE_GELU) {
          v = 0.5f * v * (1.0f + erff(v * 0.70710678118654752f));
          outB[(size_t)m * ldo + n] = (unsigned short)f2bf(v);
        } else if (mode == MODE_RESF32) {
          float r0 = resid ? resid[(size_t)m * ldo + n] : 0.f;
          outF[(size_t)m * ldo + n] = r0 + v;
        } else if (mode == MODE_QKV) {
          int b = m / Sp, s = m % Sp;
          int which = n / DIM, rem = n % DIM;
          int qh = rem >> 6, hd = rem & 63;
          int bh = b * NH + qh;
          unsigned short bv = (unsigned short)f2bf(v);
          if (which == 0)      qb [((size_t)bh * Sp + s) * HD + hd] = bv;
          else if (which == 1) kbf[((size_t)bh * Sp + s) * HD + hd] = bv;
          else                 vtb[((size_t)bh * HD + hd) * Sp + s] = bv;  // V^T
        } else { // MODE_PATCH
          int b = m / NPATCH, pi = m % NPATCH;
          v += pos[(size_t)(1 + pi) * DIM + n];
          outF[((size_t)b * SP1 + 1 + pi) * DIM + n] = v;
        }
      }
}

// ---------------------------------------------------------------------------
// Attention for one (batch,head): block = 4 waves, each wave owns 16 q-rows.
// Q natural [bh,s,64] -> A frags; K natural [bh,s,64] -> score B frags
// (contiguous along hd); V transposed [bh,64,s] -> AV B frags (contiguous
// along key). Probabilities transposed C->A via per-wave LDS.
// ---------------------------------------------------------------------------
template <int NT>
__global__ void __launch_bounds__(128, 1)
k_attn(const unsigned short* __restrict__ qb,
       const unsigned short* __restrict__ kbf,
       const unsigned short* __restrict__ vtb,
       const float* __restrict__ amask, int amstr,
       unsigned short* __restrict__ ob) {
  constexpr int Sp  = NT * 16;
  constexpr int PST = Sp + 8;                       // stagger, keeps 16B align
  __shared__ __align__(16) unsigned short P[4][16][PST];

  const int tid = threadIdx.x, lane = tid & 31, wave = tid >> 5;
  const int lh = lane >> 4, l16 = lane & 15;
  const int bh = blockIdx.x, b = bh / NH, hh = bh % NH;
  const int q0 = blockIdx.y * 64 + wave * 16;
  if (q0 >= Sp) return;                              // wave-uniform exit

  const unsigned short* qrow = qb + ((size_t)bh * Sp + q0 + l16) * HD;
  FragU qa0, qa1;
  qa0.u[0] = *(const u32x4*)(qrow + lh * 8);
  qa0.u[1] = *(const u32x4*)(qrow + 16 + lh * 8);
  qa1.u[0] = *(const u32x4*)(qrow + 32 + lh * 8);
  qa1.u[1] = *(const u32x4*)(qrow + 48 + lh * 8);

  const v8f vzero = {0.f,0.f,0.f,0.f,0.f,0.f,0.f,0.f};
  v8f sc[NT];
#pragma unroll
  for (int kt = 0; kt < NT; kt++) {
    const unsigned short* krow = kbf + ((size_t)bh * Sp + kt * 16 + l16) * HD;
    FragU k0, k1;
    k0.u[0] = *(const u32x4*)(krow + lh * 8);
    k0.u[1] = *(const u32x4*)(krow + 16 + lh * 8);
    k1.u[0] = *(const u32x4*)(krow + 32 + lh * 8);
    k1.u[1] = *(const u32x4*)(krow + 48 + lh * 8);
    v8f s = vzero;
    s = wmma_bf16(qa0, k0, s);
    s = wmma_bf16(qa1, k1, s);
    float am = amask[b * amstr + kt * 16 + l16];     // per-key additive mask
#pragma unroll
    for (int r = 0; r < 8; r++) s[r] = s[r] * 0.125f + am;  // HD^-0.5 = 1/8
    sc[kt] = s;
  }

  // row-wise softmax: row elements live across the 16 lanes of a half.
#pragma unroll
  for (int r = 0; r < 8; r++) {
    float m = -3.0e38f;
#pragma unroll
    for (int kt = 0; kt < NT; kt++) m = fmaxf(m, sc[kt][r]);
#pragma unroll
    for (int off = 1; off < 16; off <<= 1) m = fmaxf(m, __shfl_xor(m, off, 32));
    float ssum = 0.f;
#pragma unroll
    for (int kt = 0; kt < NT; kt++) {
      float e = __expf(sc[kt][r] - m);
      sc[kt][r] = e;
      ssum += e;
    }
#pragma unroll
    for (int off = 1; off < 16; off <<= 1) ssum += __shfl_xor(ssum, off, 32);
    float inv = 1.0f / ssum;
#pragma unroll
    for (int kt = 0; kt < NT; kt++) sc[kt][r] *= inv;
  }

  // transpose probs C-layout -> A-layout via wave-private LDS
#pragma unroll
  for (int kt = 0; kt < NT; kt++) {
    int col = kt * 16 + l16;
#pragma unroll
    for (int r = 0; r < 8; r++)
      P[wave][lh * 8 + r][col] = (unsigned short)f2bf(sc[kt][r]);
  }

  // o = P @ V : K-dim = keys, 4 output tiles of 16 head-dims
#pragma unroll
  for (int ot = 0; ot < 4; ot++) {
    v8f o = vzero;
    const int hd = ot * 16 + l16;
    const unsigned short* vrow = vtb + ((size_t)bh * HD + hd) * Sp;
#pragma unroll
    for (int kk = 0; kk < Sp; kk += 32) {
      FragU pa, vb;
      pa.u[0] = *(const u32x4*)&P[wave][l16][kk + lh * 8];
      pa.u[1] = *(const u32x4*)&P[wave][l16][kk + 16 + lh * 8];
      vb.u[0] = *(const u32x4*)(vrow + kk + lh * 8);
      vb.u[1] = *(const u32x4*)(vrow + kk + 16 + lh * 8);
      o = wmma_bf16(pa, vb, o);
    }
#pragma unroll
    for (int r = 0; r < 8; r++) {
      int qr = q0 + lh * 8 + r;
      ob[((size_t)b * Sp + qr) * DIM + hh * HD + ot * 16 + l16] =
          (unsigned short)f2bf(o[r]);
    }
  }
}

// ---------------------------------------------------------------------------
// LayerNorm per row (768), out bf16. rstride = input row stride.
// ---------------------------------------------------------------------------
__global__ void k_ln(const float* __restrict__ x, const float* __restrict__ g,
                     const float* __restrict__ bb, unsigned short* __restrict__ out,
                     int rstride) {
  const int row = blockIdx.x, t = threadIdx.x;
  const float* xr = x + (size_t)row * rstride;
  float v0[3], s = 0.f, ss = 0.f;
#pragma unroll
  for (int k = 0; k < 3; k++) {
    float v = xr[t + k * 256];
    v0[k] = v; s += v; ss += v * v;
  }
  __shared__ float rs[256], rq[256];
  rs[t] = s; rq[t] = ss; __syncthreads();
  for (int o = 128; o > 0; o >>= 1) {
    if (t < o) { rs[t] += rs[t + o]; rq[t] += rq[t + o]; }
    __syncthreads();
  }
  float mean = rs[0] * (1.f / 768.f);
  float var  = rq[0] * (1.f / 768.f) - mean * mean;
  float inv  = rsqrtf(var + 1e-6f);
#pragma unroll
  for (int k = 0; k < 3; k++) {
    int c = t + k * 256;
    out[(size_t)row * DIM + c] =
        (unsigned short)f2bf((v0[k] - mean) * inv * g[c] + bb[c]);
  }
}

// token squared-norms (monotone in ||x||, ranks identical)
__global__ void k_scores(const float* __restrict__ x, float* __restrict__ sco) {
  const int row = blockIdx.x, b = row / SREAL, s = row % SREAL;
  const float* xr = x + ((size_t)b * SP1 + s) * DIM;
  float ss = 0.f;
  for (int k = threadIdx.x; k < DIM; k += 256) { float v = xr[k]; ss += v * v; }
  __shared__ float rq[256];
  rq[threadIdx.x] = ss; __syncthreads();
  for (int o = 128; o > 0; o >>= 1) {
    if (threadIdx.x < o) rq[threadIdx.x] += rq[threadIdx.x + o];
    __syncthreads();
  }
  if (threadIdx.x == 0) sco[row] = rq[0];
}

// exact top-k with lower-index tie-break; CLS forced kept; stable compaction
__global__ void k_prune(const float* __restrict__ sco, float* __restrict__ am2,
                        int* __restrict__ idxmap) {
  const int b = blockIdx.x, t = threadIdx.x;
  __shared__ float s[SREAL];
  __shared__ int keep[SREAL];
  __shared__ int nk;
  if (t < SREAL) s[t] = sco[b * SREAL + t];
  __syncthreads();
  int kf = 0;
  if (t < SREAL) {
    float st = s[t];
    int cnt = 0;
    for (int j = 0; j < SREAL; j++) {
      float sj = s[j];
      cnt += (sj > st) || (sj == st && j < t);
    }
    kf = (cnt < NKEEP) || (t == 0);
    keep[t] = kf;
  }
  __syncthreads();
  if (t == 0) {
    int c = 0;
    for (int j = 0; j < SREAL; j++) c += keep[j];
    nk = c;
  }
  __syncthreads();
  if (t < SREAL && kf) {
    int pos = 0;
    for (int j = 0; j < t; j++) pos += keep[j];
    idxmap[b * SP2 + pos] = t;
  }
  if (t < SP2) {
    am2[b * SP2 + t] = (t < nk) ? 0.f : NEGM;
    if (t >= nk) idxmap[b * SP2 + t] = -1;
  }
}

__global__ void k_gather(const float* __restrict__ x, const int* __restrict__ idxmap,
                         float* __restrict__ x2) {
  const int rp = blockIdx.x, b = rp / SP2;
  const int srcv = idxmap[rp];
  const float* src = x + ((size_t)b * SP1 + (srcv < 0 ? 0 : srcv)) * DIM;
  float* dst = x2 + (size_t)rp * DIM;
  for (int k = threadIdx.x; k < DIM; k += 256)
    dst[k] = (srcv >= 0) ? src[k] : 0.f;
}

// im2col for 16x16 non-overlapping patches -> bf16 A matrix [B*196, 768]
__global__ void k_patchify(const float* __restrict__ img, unsigned short* __restrict__ ap) {
  const int idx = blockIdx.x * 256 + threadIdx.x;
  if (idx >= BT * NPATCH * DIM) return;
  int m = idx / DIM, col = idx % DIM;
  int b = m / NPATCH, pi = m % NPATCH;
  int c = col >> 8, r2 = col & 255, py = r2 >> 4, px = r2 & 15;
  int gy = pi / 14, gx = pi % 14;
  float v = img[(((size_t)b * 3 + c) * 224 + gy * 16 + py) * 224 + gx * 16 + px];
  ap[idx] = (unsigned short)f2bf(v);
}

__global__ void k_cls(float* __restrict__ x, const float* __restrict__ cls,
                      const float* __restrict__ pos) {
  const int idx = blockIdx.x * 256 + threadIdx.x;
  if (idx >= BT * DIM) return;
  int b = idx / DIM, d = idx % DIM;
  x[((size_t)b * SP1) * DIM + d] = cls[d] + pos[d];
}

__global__ void k_amaskd(float* __restrict__ am) {
  const int s = blockIdx.x * 256 + threadIdx.x;
  if (s < SP1) am[s] = (s < SREAL) ? 0.f : NEGM;
}

// ---------------------------------------------------------------------------
extern "C" void kernel_launch(void* const* d_in, const int* in_sizes, int n_in,
                              void* d_out, int out_size, void* d_ws, size_t ws_size,
                              hipStream_t stream) {
  (void)in_sizes; (void)n_in; (void)out_size; (void)ws_size;
  const float* images    = (const float*)d_in[0];
  const float* patch_w   = (const float*)d_in[1];
  const float* patch_b   = (const float*)d_in[2];
  const float* cls_token = (const float*)d_in[3];
  const float* pos_embed = (const float*)d_in[4];
  const float* ln1_g     = (const float*)d_in[5];
  const float* ln1_b     = (const float*)d_in[6];
  const float* qkv_w     = (const float*)d_in[7];
  const float* qkv_b     = (const float*)d_in[8];
  const float* proj_w    = (const float*)d_in[9];
  const float* proj_b    = (const float*)d_in[10];
  const float* ln2_g     = (const float*)d_in[11];
  const float* ln2_b     = (const float*)d_in[12];
  const float* fc1_w     = (const float*)d_in[13];
  const float* fc1_b     = (const float*)d_in[14];
  const float* fc2_w     = (const float*)d_in[15];
  const float* fc2_b     = (const float*)d_in[16];
  const float* norm_g    = (const float*)d_in[17];
  const float* norm_b    = (const float*)d_in[18];
  const float* head_w    = (const float*)d_in[19];
  const float* head_b    = (const float*)d_in[20];

  char* w = (char*)d_ws;
  auto carve = [&](size_t bytes) -> void* {
    void* p = (void*)w;
    w += (bytes + 255) & ~(size_t)255;
    return p;
  };
  float*          x      = (float*)carve((size_t)BT * SP1 * DIM * 4);
  float*          x2     = (float*)carve((size_t)BT * SP2 * DIM * 4);
  unsigned short* h      = (unsigned short*)carve((size_t)BT * SP1 * DIM * 2);
  unsigned short* ffn    = (unsigned short*)carve((size_t)BT * SP1 * DFF * 2);
  unsigned short* qb     = (unsigned short*)carve((size_t)BT * NH * SP1 * HD * 2);
  unsigned short* kbf    = (unsigned short*)carve((size_t)BT * NH * SP1 * HD * 2);
  unsigned short* vtb    = (unsigned short*)carve((size_t)BT * NH * SP1 * HD * 2);
  unsigned short* ob     = (unsigned short*)carve((size_t)BT * SP1 * DIM * 2);
  unsigned short* apatch = (unsigned short*)carve((size_t)BT * NPATCH * DIM * 2);
  // bf16 weight shadows (converted once per launch)
  unsigned short* wqkv   = (unsigned short*)carve((size_t)NLAYER * 3 * DIM * DIM * 2);
  unsigned short* wproj  = (unsigned short*)carve((size_t)NLAYER * DIM * DIM * 2);
  unsigned short* wfc1   = (unsigned short*)carve((size_t)NLAYER * DFF * DIM * 2);
  unsigned short* wfc2   = (unsigned short*)carve((size_t)NLAYER * DIM * DFF * 2);
  unsigned short* wpatch = (unsigned short*)carve((size_t)DIM * DIM * 2);
  unsigned short* whead  = (unsigned short*)carve((size_t)NCLS * DIM * 2);
  float*          sco    = (float*)carve((size_t)BT * SREAL * 4);
  float*          amD    = (float*)carve((size_t)SP1 * 4);
  float*          am2    = (float*)carve((size_t)BT * SP2 * 4);
  int*            idxmap = (int*)carve((size_t)BT * SP2 * 4);
  unsigned short* clsln  = (unsigned short*)carve((size_t)BT * DIM * 2);

  // one-shot weight downcast f32 -> bf16 (weights then live in L2 per layer)
  auto cvt = [&](const float* src, unsigned short* dst, size_t n) {
    k_cvt<<<(unsigned)((n + 255) / 256), 256, 0, stream>>>(src, dst, (int)n);
  };
  cvt(patch_w, wpatch, (size_t)DIM * DIM);
  cvt(qkv_w,  wqkv,  (size_t)NLAYER * 3 * DIM * DIM);
  cvt(proj_w, wproj, (size_t)NLAYER * DIM * DIM);
  cvt(fc1_w,  wfc1,  (size_t)NLAYER * DFF * DIM);
  cvt(fc2_w,  wfc2,  (size_t)NLAYER * DIM * DFF);
  cvt(head_w, whead, (size_t)NCLS * DIM);

  // patch embed + CLS + pos embed into x (f32 residual stream), pads zero
  hipMemsetAsync(x, 0, (size_t)BT * SP1 * DIM * 4, stream);
  {
    int n = BT * NPATCH * DIM;
    k_patchify<<<(n + 255) / 256, 256, 0, stream>>>(images, apatch);
  }
  {
    dim3 g((DIM + 127) / 128, (BT * NPATCH + 127) / 128);
    k_gemm<<<g, 256, 0, stream>>>(apatch, wpatch, patch_b, BT * NPATCH, DIM, DIM,
                                  MODE_PATCH, x, nullptr, 0, nullptr, pos_embed,
                                  nullptr, nullptr, nullptr, SP1);
  }
  k_cls<<<(BT * DIM + 255) / 256, 256, 0, stream>>>(x, cls_token, pos_embed);
  k_amaskd<<<1, 256, 0, stream>>>(amD);

  float* curx = x;
  int Sp = SP1;
  const float* am = amD;
  int amstr = 0;

  for (int i = 0; i < NLAYER; i++) {
    const int M = BT * Sp;
    // LN1 -> h (bf16)
    k_ln<<<M, 256, 0, stream>>>(curx, ln1_g + i * DIM, ln1_b + i * DIM, h, DIM);
    // QKV
    {
      dim3 g((3 * DIM + 127) / 128, (M + 127) / 128);
      k_gemm<<<g, 256, 0, stream>>>(h, wqkv + (size_t)i * 3 * DIM * DIM,
                                    qkv_b + (size_t)i * 3 * DIM, M, 3 * DIM, DIM,
                                    MODE_QKV, nullptr, nullptr, 0, nullptr, nullptr,
                                    qb, kbf, vtb, Sp);
    }
    // attention
    if (Sp == SP1)
      k_attn<SP1 / 16><<<dim3(BT * NH, (SP1 + 63) / 64), 128, 0, stream>>>(
          qb, kbf, vtb, am, amstr, ob);
    else
      k_attn<SP2 / 16><<<dim3(BT * NH, (SP2 + 63) / 64), 128, 0, stream>>>(
          qb, kbf, vtb, am, amstr, ob);
    // proj + residual
    {
      dim3 g((DIM + 127) / 128, (M + 127) / 128);
      k_gemm<<<g, 256, 0, stream>>>(ob, wproj + (size_t)i * DIM * DIM,
                                    proj_b + (size_t)i * DIM, M, DIM, DIM,
                                    MODE_RESF32, curx, nullptr, DIM, curx, nullptr,
                                    nullptr, nullptr, nullptr, Sp);
    }
    // LN2 -> h
    k_ln<<<M, 256, 0, stream>>>(curx, ln2_g + i * DIM, ln2_b + i * DIM, h, DIM);
    // FC1 + GELU
    {
      dim3 g((DFF + 127) / 128, (M + 127) / 128);
      k_gemm<<<g, 256, 0, stream>>>(h, wfc1 + (size_t)i * DFF * DIM,
                                    fc1_b + (size_t)i * DFF, M, DFF, DIM,
                                    MODE_GELU, nullptr, ffn, DFF, nullptr, nullptr,
                                    nullptr, nullptr, nullptr, Sp);
    }
    // FC2 + residual
    {
      dim3 g((DIM + 127) / 128, (M + 127) / 128);
      k_gemm<<<g, 256, 0, stream>>>(ffn, wfc2 + (size_t)i * DIM * DFF,
                                    fc2_b + (size_t)i * DIM, M, DIM, DFF,
                                    MODE_RESF32, curx, nullptr, DIM, curx, nullptr,
                                    nullptr, nullptr, nullptr, Sp);
    }
    // prune after layer 4 (i == 3 done)
    if (i == 3) {
      k_scores<<<BT * SREAL, 256, 0, stream>>>(x, sco);
      k_prune<<<BT, 256, 0, stream>>>(sco, am2, idxmap);
      k_gather<<<BT * SP2, 256, 0, stream>>>(x, idxmap, x2);
      curx = x2; Sp = SP2; am = am2; amstr = SP2;
    }
  }

  // final LN on CLS rows (row 0 of each batch) + head
  k_ln<<<BT, 256, 0, stream>>>(x2, norm_g, norm_b, clsln, SP2 * DIM);
  {
    dim3 g((NCLS + 127) / 128, (BT + 127) / 128);
    k_gemm<<<g, 256, 0, stream>>>(clsln, whead, head_b, BT, NCLS, DIM,
                                  MODE_RESF32, (float*)d_out, nullptr, NCLS,
                                  nullptr, nullptr, nullptr, nullptr, nullptr, SP2);
  }
}
